// Qwen3Attention_11398843203675
// MI455X (gfx1250) — compile-verified, matrix operand order checked by reading, not run on hardware
//
#include <hip/hip_runtime.h>

// ---------------------------------------------------------------------------
// Qwen3 attention block for MI455X (gfx1250, wave32, WMMA bf16 -> f32 acc)
// All WMMA operands use contiguous per-lane loads (transposed-B storage);
// projection GEMMs use a static two-stage software pipeline (no spills).
// ---------------------------------------------------------------------------

typedef __bf16 bf16;
typedef __attribute__((ext_vector_type(16))) __bf16 v16bf;
typedef __attribute__((ext_vector_type(8)))  __bf16 v8bf;
typedef __attribute__((ext_vector_type(8)))  float  v8f;

#define T_SEQ   2048
#define HIDDEN  2048
#define N_HEADS 16
#define N_KV    8
#define HEAD_D  128
#define QKV_N   (N_HEADS * HEAD_D + 2 * N_KV * HEAD_D)   // 4096

__device__ __forceinline__ v8f wmma_bf16(v16bf a, v16bf b, v8f c) {
    return __builtin_amdgcn_wmma_f32_16x16x32_bf16(
        /*neg_a=*/false, a, /*neg_b=*/false, b,
        /*c_mod=*/(short)0, c, /*reuse_a=*/false, /*reuse_b=*/false);
}

// Contiguous per-lane operand load (row-major source, leading dim ld).
// Serves as A operand (row = m) AND as B operand when the B matrix is stored
// transposed (row = n).  ISA layout: lane l (sel=l>>4): elems[0..7] =
// K(sel*8+j), elems[8..15] = K(16+sel*8+j) -> two 16B chunks.
__device__ __forceinline__ v16bf load_row(const bf16* M, int ld, int row,
                                          int k0, int lane) {
    int sel = lane >> 4;
    const bf16* p0 = M + (size_t)row * ld + k0 + sel * 8;
    v8bf lo = *(const v8bf*)p0;
    v8bf hi = *(const v8bf*)(p0 + 16);
    v16bf a;
#pragma unroll
    for (int j = 0; j < 8; ++j) { a[j] = lo[j]; a[8 + j] = hi[j]; }
    return a;
}

// ---------------------------------------------------------------------------
// fp32 -> bf16 conversion (grid-stride)
// ---------------------------------------------------------------------------
__global__ void cvt_bf16_kernel(const float* __restrict__ src,
                                bf16* __restrict__ dst, int n) {
    int i = blockIdx.x * blockDim.x + threadIdx.x;
    int stride = gridDim.x * blockDim.x;
    for (; i < n; i += stride) dst[i] = (bf16)src[i];
}

// ---------------------------------------------------------------------------
// fp32 [R x C] -> bf16 transposed [C x R], 32x32 tiles through LDS.
// block (32, 8); grid (C/32, R/32).
// ---------------------------------------------------------------------------
__global__ void cvt_transpose_kernel(const float* __restrict__ src,
                                     bf16* __restrict__ dst, int R, int C) {
    __shared__ bf16 tile[32][33];
    int c0 = blockIdx.x * 32, r0 = blockIdx.y * 32;
    int tx = threadIdx.x, ty = threadIdx.y;
#pragma unroll
    for (int i = 0; i < 32; i += 8)
        tile[ty + i][tx] = (bf16)src[(size_t)(r0 + ty + i) * C + c0 + tx];
    __syncthreads();
#pragma unroll
    for (int i = 0; i < 32; i += 8)
        dst[(size_t)(c0 + ty + i) * R + r0 + tx] = tile[tx][ty + i];
}

// ---------------------------------------------------------------------------
// bf16 WMMA GEMM (NT):  C[M x N](f32) = A[M x K](lda) * BT[N x K](ldb)^T
// One wave computes a 32x64 block: 2 A ops + 4 B ops -> 8 WMMAs / K-step.
// Static two-stage pipeline (cur/nxt named register sets, K-loop unrolled
// x2, K % 64 == 0) so loads of step k+32 overlap WMMAs of step k with no
// dynamic indexing and no spills.  8 waves / 256-thread block.
// ---------------------------------------------------------------------------
__global__ void __launch_bounds__(256, 1)
gemm_bf16_nt(const bf16* __restrict__ A, int lda,
             const bf16* __restrict__ BT, int ldb,
             float* __restrict__ C, int ldc,
             int M, int N, int K) {
    int lane = threadIdx.x & 31;
    int wave = blockIdx.x * (blockDim.x >> 5) + (threadIdx.x >> 5);
    int ntx = N >> 6;                       // 64-wide N tiles
    int tx = wave % ntx;
    int ty = wave / ntx;
    if (ty >= (M >> 5)) return;
    int mbase = ty << 5, nbase = tx << 6;
    int l15 = lane & 15, g = lane >> 4;

    const bf16* arow0 = A  + (size_t)(mbase + l15)      * lda;
    const bf16* arow1 = A  + (size_t)(mbase + 16 + l15) * lda;
    const bf16* brow0 = BT + (size_t)(nbase + l15)      * ldb;
    const bf16* brow1 = BT + (size_t)(nbase + 16 + l15) * ldb;
    const bf16* brow2 = BT + (size_t)(nbase + 32 + l15) * ldb;
    const bf16* brow3 = BT + (size_t)(nbase + 48 + l15) * ldb;

    v8f acc[2][4];
#pragma unroll
    for (int mi = 0; mi < 2; ++mi)
#pragma unroll
        for (int ni = 0; ni < 4; ++ni) acc[mi][ni] = v8f{};

    v16bf a0c, a1c, b0c, b1c, b2c, b3c;     // current stage
    v16bf a0n, a1n, b0n, b1n, b2n, b3n;     // next stage

#define LOAD_SET(S, K0)                                                    \
    do {                                                                   \
        a0##S = load_row(arow0, 0, 0, (K0), lane);                         \
        a1##S = load_row(arow1, 0, 0, (K0), lane);                         \
        b0##S = load_row(brow0, 0, 0, (K0), lane);                         \
        b1##S = load_row(brow1, 0, 0, (K0), lane);                         \
        b2##S = load_row(brow2, 0, 0, (K0), lane);                         \
        b3##S = load_row(brow3, 0, 0, (K0), lane);                         \
    } while (0)

#define MMA_SET(S)                                                         \
    do {                                                                   \
        acc[0][0] = wmma_bf16(a0##S, b0##S, acc[0][0]);                    \
        acc[1][0] = wmma_bf16(a1##S, b0##S, acc[1][0]);                    \
        acc[0][1] = wmma_bf16(a0##S, b1##S, acc[0][1]);                    \
        acc[1][1] = wmma_bf16(a1##S, b1##S, acc[1][1]);                    \
        acc[0][2] = wmma_bf16(a0##S, b2##S, acc[0][2]);                    \
        acc[1][2] = wmma_bf16(a1##S, b2##S, acc[1][2]);                    \
        acc[0][3] = wmma_bf16(a0##S, b3##S, acc[0][3]);                    \
        acc[1][3] = wmma_bf16(a1##S, b3##S, acc[1][3]);                    \
    } while (0)

    LOAD_SET(c, 0);
    for (int k0 = 0; k0 < K; k0 += 64) {    // K % 64 == 0 for all calls
        LOAD_SET(n, k0 + 32);               // overlap with cur WMMAs
        MMA_SET(c);
        if (k0 + 64 < K) LOAD_SET(c, k0 + 64);
        MMA_SET(n);
    }
#undef LOAD_SET
#undef MMA_SET

#pragma unroll
    for (int mi = 0; mi < 2; ++mi)
#pragma unroll
        for (int ni = 0; ni < 4; ++ni)
#pragma unroll
            for (int r = 0; r < 8; ++r) {
                int m = mbase + mi * 16 + r + g * 8;
                C[(size_t)m * ldc + nbase + ni * 16 + l15] = acc[mi][ni][r];
            }
}

// ---------------------------------------------------------------------------
// Fused RMSNorm (Q,K) + NeoX RoPE + layout + bf16 cast.
// grid: (T, 32) -> y: 0..15 Q heads, 16..23 K heads, 24..31 V heads
// Writes: qn [h][t][d]; kn [kh][t][d] (row-major: B operand of Q.K^T indexes
// lanes by kv-row); vT [kh][d][t] (transposed: B operand of P.V indexes
// lanes by d).
// ---------------------------------------------------------------------------
__global__ void norm_rope_kernel(const float* __restrict__ qkv,
                                 const float* __restrict__ q_norm_w,
                                 const float* __restrict__ k_norm_w,
                                 const int* __restrict__ positions,
                                 bf16* __restrict__ qn,
                                 bf16* __restrict__ kn,
                                 bf16* __restrict__ vT) {
    int t = blockIdx.x;
    int hh = blockIdx.y;
    int d = threadIdx.x;

    if (hh >= 24) {                                  // V heads: cast+transpose
        int kh = hh - 24;
        float v = qkv[(size_t)t * QKV_N + 3072 + kh * HEAD_D + d];
        vT[(size_t)kh * HEAD_D * T_SEQ + (size_t)d * T_SEQ + t] = (bf16)v;
        return;
    }

    __shared__ float red[HEAD_D];
    bool is_q = hh < 16;
    int colbase = is_q ? hh * HEAD_D : 2048 + (hh - 16) * HEAD_D;
    const float* w = is_q ? q_norm_w : k_norm_w;

    float x = qkv[(size_t)t * QKV_N + colbase + d];
    red[d] = x * x;
    __syncthreads();
#pragma unroll
    for (int s = 64; s > 0; s >>= 1) {
        if (d < s) red[d] += red[d + s];
        __syncthreads();
    }
    float rn = rsqrtf(red[0] * (1.0f / HEAD_D) + 1e-6f);

    int partner = d ^ 64;                            // rotate-half pair
    float xp = qkv[(size_t)t * QKV_N + colbase + partner];
    float xn  = x  * rn * w[d];
    float xpn = xp * rn * w[partner];

    int i = d & 63;
    float inv_freq = powf(1000000.0f, -((float)(2 * i)) / (float)HEAD_D);
    float fr = (float)positions[t] * inv_freq;
    float c = cosf(fr), s = sinf(fr);
    float out = (d < 64) ? (xn * c - xpn * s) : (xn * c + xpn * s);

    if (is_q) {
        qn[(size_t)hh * T_SEQ * HEAD_D + (size_t)t * HEAD_D + d] = (bf16)out;
    } else {
        int kh = hh - 16;
        kn[(size_t)kh * T_SEQ * HEAD_D + (size_t)t * HEAD_D + d] = (bf16)out;
    }
}

// ---------------------------------------------------------------------------
// Causal flash attention, one wave per 16-row Q tile, kv tiles of 32.
// grid: (T/16/4, N_HEADS), block: 128 (4 waves).
// ---------------------------------------------------------------------------
__global__ void flash_attn_kernel(const bf16* __restrict__ qn,
                                  const bf16* __restrict__ kn,
                                  const bf16* __restrict__ vT,
                                  bf16* __restrict__ attn_out) {
    int lane = threadIdx.x & 31;
    int wid  = threadIdx.x >> 5;
    int h = blockIdx.y;
    int qtile = blockIdx.x * 4 + wid;
    int qbase = qtile * 16;
    int kvh = h >> 1;                                // GQA: 2 Q heads / KV head

    const bf16* Q  = qn + (size_t)h   * T_SEQ * HEAD_D;
    const bf16* Kh = kn + (size_t)kvh * T_SEQ * HEAD_D;   // [t][d]
    const bf16* Vt = vT + (size_t)kvh * HEAD_D * T_SEQ;   // [d][t]

    __shared__ __align__(16) bf16 pstage[4][16 * 32];
    bf16* P = pstage[wid];

    int l15 = lane & 15, g = lane >> 4;

    // Q tile resident in registers: 16 x 128 as four A operands
    v16bf aq[4];
#pragma unroll
    for (int c = 0; c < 4; ++c)
        aq[c] = load_row(Q, HEAD_D, qbase + l15, c * 32, lane);

    v8f o[8];
#pragma unroll
    for (int i = 0; i < 8; ++i) o[i] = v8f{};
    float mrow[8], lrow[8];
#pragma unroll
    for (int r = 0; r < 8; ++r) { mrow[r] = -1e30f; lrow[r] = 0.0f; }

    const float scale = 0.08838834764831845f;        // 1/sqrt(128)
    int jend = qbase + 16;                           // causal kv extent

    for (int j0 = 0; j0 < jend; j0 += 32) {
        // ---- S = Q . K^T : B operand = K rows (lane n = kv-row) ----------
        v8f s0 = {}, s1 = {};
#pragma unroll
        for (int c = 0; c < 4; ++c) {
            v16bf b0 = load_row(Kh, HEAD_D, j0 + l15,      c * 32, lane);
            s0 = wmma_bf16(aq[c], b0, s0);
            v16bf b1 = load_row(Kh, HEAD_D, j0 + 16 + l15, c * 32, lane);
            s1 = wmma_bf16(aq[c], b1, s1);
        }

        int col0 = j0 + l15;
        int col1 = col0 + 16;

        // ---- online softmax (per-row stats replicated over 16-lane half) --
#pragma unroll
        for (int r = 0; r < 8; ++r) {
            int rr = qbase + r + g * 8;
            float sm0 = (col0 <= rr) ? s0[r] * scale : -1e30f;
            float sm1 = (col1 <= rr) ? s1[r] * scale : -1e30f;
            float mx = fmaxf(sm0, sm1);
#pragma unroll
            for (int dlt = 1; dlt < 16; dlt <<= 1)
                mx = fmaxf(mx, __shfl_xor(mx, dlt, 32));
            float mnew = fmaxf(mrow[r], mx);
            float corr = __expf(mrow[r] - mnew);
            float p0 = __expf(sm0 - mnew);
            float p1 = __expf(sm1 - mnew);
            float ps = p0 + p1;
#pragma unroll
            for (int dlt = 1; dlt < 16; dlt <<= 1)
                ps += __shfl_xor(ps, dlt, 32);
            lrow[r] = lrow[r] * corr + ps;
            mrow[r] = mnew;
#pragma unroll
            for (int dt = 0; dt < 8; ++dt) o[dt][r] *= corr;
            // stage P (C-layout -> LDS row-major 16x32)
            int prow = r + g * 8;
            P[prow * 32 +      l15] = (bf16)p0;
            P[prow * 32 + 16 + l15] = (bf16)p1;
        }

        // wave-synchronous LDS RAW: drain DS ops before A-layout reload
        asm volatile("s_wait_dscnt 0" ::: "memory");

        // ---- O += P . V : B operand = V^T rows (lane n = d) --------------
        v16bf ap = load_row(P, 32, l15, 0, lane);
#pragma unroll
        for (int dt = 0; dt < 8; ++dt) {
            v16bf bv = load_row(Vt, T_SEQ, dt * 16 + l15, j0, lane);
            o[dt] = wmma_bf16(ap, bv, o[dt]);
        }
    }

    // ---- finalize: divide by l, cast bf16, write [t][h*128 + d] ----------
#pragma unroll
    for (int r = 0; r < 8; ++r) {
        float inv = 1.0f / lrow[r];
        int row = qbase + r + g * 8;
#pragma unroll
        for (int dt = 0; dt < 8; ++dt) {
            float val = o[dt][r] * inv;
            attn_out[(size_t)row * (N_HEADS * HEAD_D) + h * HEAD_D +
                     dt * 16 + l15] = (bf16)val;
        }
    }
}

// ---------------------------------------------------------------------------
// Host-side orchestration
// ---------------------------------------------------------------------------
extern "C" void kernel_launch(void* const* d_in, const int* in_sizes, int n_in,
                              void* d_out, int out_size, void* d_ws, size_t ws_size,
                              hipStream_t stream) {
    const int*   positions = (const int*)  d_in[0];
    const float* hidden    = (const float*)d_in[1];
    const float* wq        = (const float*)d_in[2];
    const float* wk        = (const float*)d_in[3];
    const float* wv        = (const float*)d_in[4];
    const float* wo        = (const float*)d_in[5];
    const float* q_norm_w  = (const float*)d_in[6];
    const float* k_norm_w  = (const float*)d_in[7];
    float* out = (float*)d_out;

    char* ws = (char*)d_ws;
    size_t off = 0;
    auto carve = [&](size_t bytes) -> void* {
        void* p = ws + off;
        off += (bytes + 255) & ~(size_t)255;
        return p;
    };

    bf16*  hs_b   = (bf16*) carve((size_t)T_SEQ * HIDDEN * 2);    // [T][K]
    bf16*  wqT    = (bf16*) carve((size_t)2048 * HIDDEN * 2);     // [N][K]
    bf16*  wkT    = (bf16*) carve((size_t)1024 * HIDDEN * 2);
    bf16*  wvT    = (bf16*) carve((size_t)1024 * HIDDEN * 2);
    bf16*  woT    = (bf16*) carve((size_t)HIDDEN * 2048 * 2);
    float* qkv_f  = (float*)carve((size_t)T_SEQ * QKV_N * 4);
    bf16*  qn_b   = (bf16*) carve((size_t)N_HEADS * T_SEQ * HEAD_D * 2);
    bf16*  kn_b   = (bf16*) carve((size_t)N_KV * T_SEQ * HEAD_D * 2);
    bf16*  vT_b   = (bf16*) carve((size_t)N_KV * HEAD_D * T_SEQ * 2);
    bf16*  attn_b = (bf16*) carve((size_t)T_SEQ * N_HEADS * HEAD_D * 2);

    // 1) activations: straight cast; weights: cast + transpose to [N][K]
    cvt_bf16_kernel<<<2048, 256, 0, stream>>>(hidden, hs_b, T_SEQ * HIDDEN);
    cvt_transpose_kernel<<<dim3(2048 / 32, HIDDEN / 32), dim3(32, 8), 0, stream>>>(
        wq, wqT, HIDDEN, 2048);
    cvt_transpose_kernel<<<dim3(1024 / 32, HIDDEN / 32), dim3(32, 8), 0, stream>>>(
        wk, wkT, HIDDEN, 1024);
    cvt_transpose_kernel<<<dim3(1024 / 32, HIDDEN / 32), dim3(32, 8), 0, stream>>>(
        wv, wvT, HIDDEN, 1024);
    cvt_transpose_kernel<<<dim3(HIDDEN / 32, 2048 / 32), dim3(32, 8), 0, stream>>>(
        wo, woT, 2048, HIDDEN);

    // 2) QKV projections into fused [T x 4096] f32 buffer (NT GEMMs)
    {   // Q: M=2048 N=2048 K=2048 -> waves = 64*32
        int blocks = (T_SEQ / 32) * (2048 / 64) / 8;
        gemm_bf16_nt<<<blocks, 256, 0, stream>>>(
            hs_b, HIDDEN, wqT, HIDDEN, qkv_f + 0, QKV_N, T_SEQ, 2048, HIDDEN);
    }
    {   // K: N=1024
        int blocks = (T_SEQ / 32) * (1024 / 64) / 8;
        gemm_bf16_nt<<<blocks, 256, 0, stream>>>(
            hs_b, HIDDEN, wkT, HIDDEN, qkv_f + 2048, QKV_N, T_SEQ, 1024, HIDDEN);
    }
    {   // V: N=1024
        int blocks = (T_SEQ / 32) * (1024 / 64) / 8;
        gemm_bf16_nt<<<blocks, 256, 0, stream>>>(
            hs_b, HIDDEN, wvT, HIDDEN, qkv_f + 3072, QKV_N, T_SEQ, 1024, HIDDEN);
    }

    // 3) RMSNorm + RoPE + layout (Q,K row-major; V transposed)
    norm_rope_kernel<<<dim3(T_SEQ, 32), 128, 0, stream>>>(
        qkv_f, q_norm_w, k_norm_w, positions, qn_b, kn_b, vT_b);

    // 4) causal flash attention (online softmax, WMMA QK^T and PV)
    flash_attn_kernel<<<dim3((T_SEQ / 16) / 4, N_HEADS), 128, 0, stream>>>(
        qn_b, kn_b, vT_b, attn_b);

    // 5) output projection: [2048 x 2048] @ wo -> f32 out
    {
        int blocks = (T_SEQ / 32) * (HIDDEN / 64) / 8;
        gemm_bf16_nt<<<blocks, 256, 0, stream>>>(
            attn_b, N_HEADS * HEAD_D, woT, HIDDEN, out, HIDDEN,
            T_SEQ, HIDDEN, N_HEADS * HEAD_D);
    }
    (void)in_sizes; (void)n_in; (void)out_size; (void)ws_size;
}
